// CoAttentionSonLayer_74912819577426
// MI455X (gfx1250) — compile-verified, matrix operand order checked by reading, not run on hardware
//
#include <hip/hip_runtime.h>
#include <hip/hip_bf16.h>

typedef __attribute__((ext_vector_type(16))) _Float16 v16h;
typedef __attribute__((ext_vector_type(8)))  _Float16 v8h;
typedef __attribute__((ext_vector_type(8)))  float    v8f;

#define B_SZ  4
#define SQ_   2048
#define SKV_  2048
#define H_    1024
#define NH_   16
#define HD_   64

// ---------------------------------------------------------------------------
// f32 -> f16 conversion
// ---------------------------------------------------------------------------
__global__ __launch_bounds__(256) void cvt_f32_f16(const float* __restrict__ in,
                                                   _Float16* __restrict__ out, int n) {
  int i = blockIdx.x * blockDim.x + threadIdx.x;
  if (i < n) out[i] = (_Float16)in[i];
}

// ---------------------------------------------------------------------------
// Fragment loader: A-frag (16x32, MxK) and B-frag (32x16, KxN read as N rows
// of K, i.e. W row-major) share the same per-lane layout:
//   idx16 = lane & 15  (M row / N col), koff = (lane>>4)*8
//   elems [0..7]  = base[(row)*stride + k0 + koff + t]
//   elems [8..15] = base[(row)*stride + k0 + 16 + koff + t]
// All addresses are 16-byte aligned (k0 % 32 == 0, koff in {0,8}).
// ---------------------------------------------------------------------------
__device__ __forceinline__ v16h load_frag(const _Float16* __restrict__ base,
                                          int stride, int row16, int k0, int lane) {
  int r    = row16 + (lane & 15);
  int koff = (lane >> 4) * 8;
  const _Float16* p = base + (size_t)r * stride + k0 + koff;
  v8h lo = *(const v8h*)(p);
  v8h hi = *(const v8h*)(p + 16);
  v16h a;
#pragma unroll
  for (int t = 0; t < 8; ++t) { a[t] = lo[t]; a[8 + t] = hi[t]; }
  return a;
}

// ---------------------------------------------------------------------------
// WMMA GEMM: out[m,n] = sum_k A[m,k] * W[n,k] (+ bias[n]) (+ residual)
//   mode 0: out16[m*N + n]                             (Q, K projections)
//   mode 1: out16[((b*NH+h)*HD+d)*S + s]  (V, transposed per-head)
//   mode 2: out32[m*N + n] = acc + bias + residual     (O projection)
// 4 waves per block arranged 2x2; each wave owns a 64x64 tile (4x4 WMMA
// fragments -> 16 wmma per 16 b128 loads per K=32 step); block = 128x128.
// ---------------------------------------------------------------------------
__global__ __launch_bounds__(128) void gemm_wmma_f16(
    const _Float16* __restrict__ A,   // [M,K]
    const _Float16* __restrict__ W,   // [N,K]
    const float*    __restrict__ bias,// [N] or null
    _Float16*       __restrict__ out16,
    float*          __restrict__ out32,
    const float*    __restrict__ residual,
    int M, int N, int K, int mode, int S) {
  int lane = threadIdx.x & 31;
  int wave = threadIdx.x >> 5;
  int m0 = blockIdx.y * 128 + (wave >> 1) * 64;
  int n0 = blockIdx.x * 128 + (wave & 1) * 64;

  v8f acc[4][4] = {};
  for (int k0 = 0; k0 < K; k0 += 32) {
    v16h a[4], b[4];
#pragma unroll
    for (int i = 0; i < 4; ++i) a[i] = load_frag(A, K, m0 + 16 * i, k0, lane);
#pragma unroll
    for (int j = 0; j < 4; ++j) b[j] = load_frag(W, K, n0 + 16 * j, k0, lane);

    // prefetch the K-slice two iterations ahead (global_prefetch_b8)
    int kpf = k0 + 64;
    if (kpf < K) {
      __builtin_prefetch(A + (size_t)(m0 + lane) * K + kpf, 0, 0);
      __builtin_prefetch(A + (size_t)(m0 + 32 + lane) * K + kpf, 0, 0);
      __builtin_prefetch(W + (size_t)(n0 + lane) * K + kpf, 0, 0);
      __builtin_prefetch(W + (size_t)(n0 + 32 + lane) * K + kpf, 0, 0);
    }

#pragma unroll
    for (int i = 0; i < 4; ++i)
#pragma unroll
      for (int j = 0; j < 4; ++j)
        acc[i][j] = __builtin_amdgcn_wmma_f32_16x16x32_f16(false, a[i], false, b[j],
                                                           (short)0, acc[i][j], false, false);
  }

  int half = lane >> 4;
  int nc   = lane & 15;
#pragma unroll
  for (int i = 0; i < 4; ++i)
#pragma unroll
    for (int j = 0; j < 4; ++j) {
      int col = n0 + 16 * j + nc;
      float bcol = bias ? bias[col] : 0.0f;
#pragma unroll
      for (int v = 0; v < 8; ++v) {
        int row = m0 + 16 * i + 8 * half + v;
        float val = acc[i][j][v] + bcol;
        if (mode == 0) {
          out16[(size_t)row * N + col] = (_Float16)val;
        } else if (mode == 1) {
          int bb = row / S, sv = row % S;
          int hh = col / HD_, dd = col % HD_;
          out16[(((size_t)bb * NH_ + hh) * HD_ + dd) * S + sv] = (_Float16)val;
        } else {
          size_t idx = (size_t)row * N + col;
          out32[idx] = val + residual[idx];
        }
      }
    }
}

// ---------------------------------------------------------------------------
// Flash attention: per (b,h) and 32-row q tile (one wave each, 4 waves/block).
//   S = Q K^T * 1/sqrt(HD) via WMMA, online softmax (row stats reduced with
//   shfl_xor across the 16-lane half holding each row), P staged through LDS
//   (C-layout -> A-layout transpose), P @ V via WMMA (V stored [b,h,HD,SKV]).
// ---------------------------------------------------------------------------
__global__ __launch_bounds__(128) void attn_wmma(
    const _Float16* __restrict__ Q,   // [B,SQ,H]  (per-head stride H)
    const _Float16* __restrict__ Km,  // [B,SKV,H]
    const _Float16* __restrict__ Vt,  // [B,NH,HD,SKV]
    _Float16* __restrict__ ctx) {     // [B,SQ,H]
  int lane = threadIdx.x & 31;
  int wave = threadIdx.x >> 5;
  int bh = blockIdx.y;
  int b = bh / NH_, h = bh % NH_;
  int q0 = blockIdx.x * 128 + wave * 32;

  __shared__ __align__(16) _Float16 lds[4][32][40];  // P tile per wave, padded rows

  const _Float16* Qb = Q  + ((size_t)b * SQ_) * H_ + h * HD_;
  const _Float16* Kb = Km + ((size_t)b * SKV_) * H_ + h * HD_;
  const _Float16* Vb = Vt + (((size_t)b * NH_ + h) * HD_) * SKV_;

  // Q fragments held in registers for the whole kv loop
  v16h qa[2][2];
#pragma unroll
  for (int i = 0; i < 2; ++i)
#pragma unroll
    for (int ks = 0; ks < 2; ++ks)
      qa[i][ks] = load_frag(Qb, H_, q0 + 16 * i, ks * 32, lane);

  v8f acc[2][4] = {};
  float rowmax[2][8], rowsum[2][8];
#pragma unroll
  for (int i = 0; i < 2; ++i)
#pragma unroll
    for (int v = 0; v < 8; ++v) { rowmax[i][v] = -1e30f; rowsum[i][v] = 0.0f; }

  const float scale = 0.125f;  // 1/sqrt(64)
  int half = lane >> 4;
  int nc   = lane & 15;

  for (int kv0 = 0; kv0 < SKV_; kv0 += 32) {
    // ---- S = Q K^T ----
    v16h kb[2][2];
#pragma unroll
    for (int j = 0; j < 2; ++j)
#pragma unroll
      for (int ks = 0; ks < 2; ++ks)
        kb[j][ks] = load_frag(Kb, H_, kv0 + 16 * j, ks * 32, lane);

    v8f s[2][2] = {};
#pragma unroll
    for (int i = 0; i < 2; ++i)
#pragma unroll
      for (int j = 0; j < 2; ++j)
#pragma unroll
        for (int ks = 0; ks < 2; ++ks)
          s[i][j] = __builtin_amdgcn_wmma_f32_16x16x32_f16(false, qa[i][ks], false, kb[j][ks],
                                                           (short)0, s[i][j], false, false);

    // ---- online softmax ----
#pragma unroll
    for (int i = 0; i < 2; ++i) {
      float tmax[8];
#pragma unroll
      for (int v = 0; v < 8; ++v) {
        s[i][0][v] *= scale;
        s[i][1][v] *= scale;
        tmax[v] = fmaxf(s[i][0][v], s[i][1][v]);
      }
#pragma unroll
      for (int off = 1; off < 16; off <<= 1)
#pragma unroll
        for (int v = 0; v < 8; ++v)
          tmax[v] = fmaxf(tmax[v], __shfl_xor(tmax[v], off, 32));

      float tsum[8];
#pragma unroll
      for (int v = 0; v < 8; ++v) {
        float nm   = fmaxf(rowmax[i][v], tmax[v]);
        float corr = __expf(rowmax[i][v] - nm);
        rowmax[i][v] = nm;
        rowsum[i][v] *= corr;
#pragma unroll
        for (int jd = 0; jd < 4; ++jd) acc[i][jd][v] *= corr;
        float p0 = __expf(s[i][0][v] - nm);
        float p1 = __expf(s[i][1][v] - nm);
        s[i][0][v] = p0;
        s[i][1][v] = p1;
        tsum[v] = p0 + p1;
      }
#pragma unroll
      for (int off = 1; off < 16; off <<= 1)
#pragma unroll
        for (int v = 0; v < 8; ++v)
          tsum[v] += __shfl_xor(tsum[v], off, 32);
#pragma unroll
      for (int v = 0; v < 8; ++v) rowsum[i][v] += tsum[v];

      // P tile (C layout) -> LDS
#pragma unroll
      for (int j = 0; j < 2; ++j)
#pragma unroll
        for (int v = 0; v < 8; ++v)
          lds[wave][16 * i + 8 * half + v][16 * j + nc] = (_Float16)s[i][j][v];
    }
    __syncthreads();

    // ---- P (A layout from LDS) @ V ----
    v16h pa[2];
#pragma unroll
    for (int i = 0; i < 2; ++i) {
      int r    = 16 * i + (lane & 15);
      int koff = (lane >> 4) * 8;
      const _Float16* lp = &lds[wave][r][0];
      v8h lo = *(const v8h*)(lp + koff);
      v8h hi = *(const v8h*)(lp + 16 + koff);
#pragma unroll
      for (int t = 0; t < 8; ++t) { pa[i][t] = lo[t]; pa[i][8 + t] = hi[t]; }
    }
    v16h vb[4];
#pragma unroll
    for (int jd = 0; jd < 4; ++jd)
      vb[jd] = load_frag(Vb, SKV_, 16 * jd, kv0, lane);  // row = d, k = kv
#pragma unroll
    for (int i = 0; i < 2; ++i)
#pragma unroll
      for (int jd = 0; jd < 4; ++jd)
        acc[i][jd] = __builtin_amdgcn_wmma_f32_16x16x32_f16(false, pa[i], false, vb[jd],
                                                            (short)0, acc[i][jd], false, false);
    __syncthreads();
  }

  // ---- normalize + store ctx ----
#pragma unroll
  for (int i = 0; i < 2; ++i)
#pragma unroll
    for (int jd = 0; jd < 4; ++jd)
#pragma unroll
      for (int v = 0; v < 8; ++v) {
        int row = q0 + 16 * i + 8 * half + v;
        int d   = 16 * jd + nc;
        float o = acc[i][jd][v] / rowsum[i][v];
        ctx[((size_t)b * SQ_ + row) * H_ + h * HD_ + d] = (_Float16)o;
      }
}

// ---------------------------------------------------------------------------
// RMSNorm: one block per row of h [B*SQ, H]
// ---------------------------------------------------------------------------
__global__ __launch_bounds__(256) void rmsnorm_kernel(const float* __restrict__ hbuf,
                                                      const float* __restrict__ gamma,
                                                      float* __restrict__ out) {
  int row = blockIdx.x;
  const float* hr = hbuf + (size_t)row * H_;
  float ss = 0.0f;
  for (int c = threadIdx.x; c < H_; c += 256) { float x = hr[c]; ss += x * x; }
#pragma unroll
  for (int off = 16; off > 0; off >>= 1) ss += __shfl_xor(ss, off, 32);
  __shared__ float red[8];
  int wave = threadIdx.x >> 5, lane = threadIdx.x & 31;
  if (lane == 0) red[wave] = ss;
  __syncthreads();
  if (wave == 0) {
    float t = (lane < 8) ? red[lane] : 0.0f;
#pragma unroll
    for (int off = 4; off > 0; off >>= 1) t += __shfl_xor(t, off, 32);
    if (lane == 0) red[0] = t;
  }
  __syncthreads();
  float rms = rsqrtf(red[0] / (float)H_ + 1e-6f);
  for (int c = threadIdx.x; c < H_; c += 256)
    out[(size_t)row * H_ + c] = hr[c] * rms * gamma[c];
}

// ---------------------------------------------------------------------------
extern "C" void kernel_launch(void* const* d_in, const int* in_sizes, int n_in,
                              void* d_out, int out_size, void* d_ws, size_t ws_size,
                              hipStream_t stream) {
  (void)in_sizes; (void)n_in; (void)out_size; (void)ws_size;
  const float* hs = (const float*)d_in[0];   // [B,SQ,H]
  const float* kv = (const float*)d_in[1];   // [B,SKV,H]
  const float* Wq = (const float*)d_in[2];
  const float* bq = (const float*)d_in[3];
  const float* Wk = (const float*)d_in[4];
  const float* bk = (const float*)d_in[5];
  const float* Wv = (const float*)d_in[6];
  const float* bv = (const float*)d_in[7];
  const float* Wo = (const float*)d_in[8];
  const float* bo = (const float*)d_in[9];
  const float* gamma = (const float*)d_in[10];
  float* out = (float*)d_out;

  const int M = B_SZ * SQ_;        // 8192
  const int nAct = M * H_;         // 8388608
  const int nW = H_ * H_;          // 1048576

  char* ws = (char*)d_ws;
  size_t off = 0;
  _Float16* hs16  = (_Float16*)(ws + off); off += (size_t)nAct * 2;
  _Float16* kv16  = (_Float16*)(ws + off); off += (size_t)nAct * 2;
  _Float16* Wq16  = (_Float16*)(ws + off); off += (size_t)nW * 2;
  _Float16* Wk16  = (_Float16*)(ws + off); off += (size_t)nW * 2;
  _Float16* Wv16  = (_Float16*)(ws + off); off += (size_t)nW * 2;
  _Float16* Wo16  = (_Float16*)(ws + off); off += (size_t)nW * 2;
  _Float16* Q16   = (_Float16*)(ws + off); off += (size_t)nAct * 2;
  _Float16* K16   = (_Float16*)(ws + off); off += (size_t)nAct * 2;
  _Float16* Vt16  = (_Float16*)(ws + off); off += (size_t)nAct * 2;
  _Float16* ctx16 = (_Float16*)(ws + off); off += (size_t)nAct * 2;
  float*    h32   = (float*)   (ws + off); off += (size_t)nAct * 4;

  // f32 -> f16 conversions
  cvt_f32_f16<<<(nAct + 255) / 256, 256, 0, stream>>>(hs, hs16, nAct);
  cvt_f32_f16<<<(nAct + 255) / 256, 256, 0, stream>>>(kv, kv16, nAct);
  cvt_f32_f16<<<(nW + 255) / 256, 256, 0, stream>>>(Wq, Wq16, nW);
  cvt_f32_f16<<<(nW + 255) / 256, 256, 0, stream>>>(Wk, Wk16, nW);
  cvt_f32_f16<<<(nW + 255) / 256, 256, 0, stream>>>(Wv, Wv16, nW);
  cvt_f32_f16<<<(nW + 255) / 256, 256, 0, stream>>>(Wo, Wo16, nW);

  // Q/K/V projections (WMMA): grid covers 128x128 tiles
  dim3 ggrid(H_ / 128, M / 128);  // (8, 64)
  gemm_wmma_f16<<<ggrid, 128, 0, stream>>>(hs16, Wq16, bq, Q16, nullptr, nullptr,
                                           M, H_, H_, 0, SKV_);
  gemm_wmma_f16<<<ggrid, 128, 0, stream>>>(kv16, Wk16, bk, K16, nullptr, nullptr,
                                           M, H_, H_, 0, SKV_);
  gemm_wmma_f16<<<ggrid, 128, 0, stream>>>(kv16, Wv16, bv, Vt16, nullptr, nullptr,
                                           M, H_, H_, 1, SKV_);

  // attention (WMMA QK^T + PV, online softmax)
  attn_wmma<<<dim3(SQ_ / 128, B_SZ * NH_), 128, 0, stream>>>(Q16, K16, Vt16, ctx16);

  // O projection + bias + residual (f32 accum out)
  gemm_wmma_f16<<<ggrid, 128, 0, stream>>>(ctx16, Wo16, bo, nullptr, h32, hs,
                                           M, H_, H_, 2, SKV_);

  // RMSNorm * gamma
  rmsnorm_kernel<<<M, 256, 0, stream>>>(h32, gamma, out);
}
